// MoE_MLP_1657857376802
// MI455X (gfx1250) — matrix operand order repaired and checked by reading, not compiled, not run on hardware
//
#include <hip/hip_runtime.h>

// ---------------- types for CDNA5 WMMA ----------------
typedef __attribute__((ext_vector_type(16))) __bf16        v16bf;
typedef __attribute__((ext_vector_type(8)))  float         v8f;
typedef __attribute__((ext_vector_type(4)))  unsigned int  u32x4;
typedef __attribute__((ext_vector_type(2)))  unsigned int  u32x2;

union FragU { u32x4 q[2]; v16bf v; };

// bf16 truncation store: lowers to ds_store_b16_d16_hi (free convert)
__device__ __forceinline__ void st_bf16_hi(unsigned short* p, float v) {
  *p = (unsigned short)(__float_as_uint(v) >> 16);
}
// pack two floats' high halves into one dword (v_perm-able)
__device__ __forceinline__ unsigned pk_bf16(float a, float b) {
  return (__float_as_uint(a) >> 16) | (__float_as_uint(b) & 0xFFFF0000u);
}
// single-instruction ReLU: v_med3_num_f32(v, 0, +inf)
__device__ __forceinline__ float relu1(float v) {
  return __builtin_amdgcn_fmed3f(v, 0.f, __builtin_inff());
}

__device__ __forceinline__ v8f wmma_bf16(v16bf a, v16bf b, v8f c) {
  // v_wmma_f32_16x16x32_bf16: (neg_a, A, neg_b, B, c_mod, C, reuse_a, reuse_b)
  return __builtin_amdgcn_wmma_f32_16x16x32_bf16(false, a, false, b, (short)0, c, false, false);
}

// A fragment (16x32 bf16): lane<16 -> row=lane, K chunks +0..7 and +16..23;
// lane>=16 -> row=lane-16, K chunks +8..15 and +24..31.  (ISA 7.12.2)
__device__ __forceinline__ v16bf load_frag_a(const unsigned short* base, int row,
                                             int stride, int k0, int sel8) {
  const unsigned short* p = base + row * stride + k0 + sel8;
  FragU f;
  f.q[0] = *(const u32x4*)(p);
  f.q[1] = *(const u32x4*)(p + 16);
  return f.v;
}
// B fragment (32x16 bf16) from W^T (rows = N, contiguous K):
// lane<16 -> col n, K 0..7 and 16..23; lane>=16 -> col n, K 8..15 and 24..31.
__device__ __forceinline__ v16bf load_frag_b(const unsigned short* base, int n,
                                             int stride, int k0, int sel16) {
  const unsigned short* p = base + n * stride + k0 + (sel16 >> 1);
  FragU f;
  f.q[0] = *(const u32x4*)(p);
  f.q[1] = *(const u32x4*)(p + 16);
  return f.v;
}

// ---------------- problem constants ----------------
#define NTOK 65536
#define DDIM 128
#define HDIM 256
#define NEXP 16
#define TILE_M 128
#define MAXTILES (NTOK / TILE_M + NEXP)   // 528

// workspace layout (ints)
#define WS_COUNTS 0
#define WS_FILL   16
#define WS_TSTART 32
#define WS_EIDX   64
#define WS_IDXBUF (64 + NTOK)

// LDS strides (halves), 16B-multiple, odd-dword for bank spread
#define XK 136
#define WK1 136
#define WK2 264
#define HK  264
#define SMEM_HALVES (128*XK + 256*WK2 + 128*HK + 128*HK)

// ---------------- kernel 0: init counters ----------------
__global__ void moe_init_kernel(int* __restrict__ wsi) {
  if (threadIdx.x < 32) wsi[threadIdx.x] = 0;
}

// ---------------- kernel 1: gate + argmax + histogram ----------------
__global__ __launch_bounds__(256)
void moe_gate_kernel(const float* __restrict__ x, const float* __restrict__ Wg,
                     const float* __restrict__ bg, int* __restrict__ wsi) {
  const int n = blockIdx.x * 256 + threadIdx.x;
  const float4* xr = (const float4*)(x + (size_t)n * DDIM);
  float acc[NEXP];
#pragma unroll
  for (int e = 0; e < NEXP; ++e) acc[e] = bg[e];
#pragma unroll 4
  for (int d4 = 0; d4 < DDIM / 4; ++d4) {
    float4 xv = xr[d4];
    const float* wr = Wg + d4 * 4 * NEXP;   // uniform -> scalar loads
#pragma unroll
    for (int e = 0; e < NEXP; ++e)
      acc[e] += xv.x * wr[e] + xv.y * wr[NEXP + e] +
                xv.z * wr[2 * NEXP + e] + xv.w * wr[3 * NEXP + e];
  }
  int bi = 0; float bv = acc[0];
#pragma unroll
  for (int e = 1; e < NEXP; ++e)
    if (acc[e] > bv) { bv = acc[e]; bi = e; }   // first-max (jnp.argmax ties)
  wsi[WS_EIDX + n] = bi;
  atomicAdd(&wsi[WS_COUNTS + bi], 1);
}

// ---------------- kernel 2: padded prefix over 16 experts ----------------
__global__ void moe_prefix_kernel(int* __restrict__ wsi) {
  if (threadIdx.x == 0) {
    int acc = 0;
    for (int e = 0; e < NEXP; ++e) {
      wsi[WS_TSTART + e] = acc;
      acc += (wsi[WS_COUNTS + e] + TILE_M - 1) >> 7;   // tiles of 128
    }
    wsi[WS_TSTART + NEXP] = acc;                       // total tiles
  }
}

// ---------------- kernel 3: scatter token ids into segments ----------------
__global__ __launch_bounds__(256)
void moe_scatter_kernel(int* __restrict__ wsi) {
  const int n = blockIdx.x * 256 + threadIdx.x;
  const int e = wsi[WS_EIDX + n];
  const int pos = atomicAdd(&wsi[WS_FILL + e], 1);
  wsi[WS_IDXBUF + wsi[WS_TSTART + e] * TILE_M + pos] = n;
}

// ---------------- kernel 4: expert MLP over 128-token tiles ----------------
__global__ __launch_bounds__(256)
void moe_mlp_kernel(const float* __restrict__ x,
                    const float* __restrict__ W1, const float* __restrict__ b1,
                    const float* __restrict__ W2, const float* __restrict__ b2,
                    const float* __restrict__ W3, const float* __restrict__ b3,
                    const int* __restrict__ wsi, float* __restrict__ out) {
  extern __shared__ unsigned short smem[];
  unsigned short* XS = smem;                  // 128 x 128 (stride XK); later W3^T row 0
  unsigned short* WS = XS + 128 * XK;         // W^T staging (<= 256 x 256)
  unsigned short* H1 = WS + 256 * WK2;        // 128 x 256
  unsigned short* H2 = H1 + 128 * HK;         // 128 x 256

  const int tid = threadIdx.x;
  const int blk = blockIdx.x;
  const int total = wsi[WS_TSTART + NEXP];
  if (blk >= total) return;                   // uniform exit, before barriers

  int e = 0;
#pragma unroll
  for (int i = 1; i < NEXP; ++i)
    if (wsi[WS_TSTART + i] <= blk) e = i;
  const int lt  = blk - wsi[WS_TSTART + e];
  const int cnt = wsi[WS_COUNTS + e];
  const int slotBase = wsi[WS_TSTART + e] * TILE_M + lt * TILE_M;
  const int* idxbuf = wsi + WS_IDXBUF;

  // ---- gather x tile -> bf16 LDS, packed dword stores (2 threads / token) ----
  {
    const int row = tid >> 1, half = tid & 1;
    const bool valid = (lt * TILE_M + row) < cnt;
    const int tok = valid ? idxbuf[slotBase + row] : 0;
    const float4* src = (const float4*)(x + (size_t)tok * DDIM + half * 64);
    unsigned short* dst = XS + row * XK + half * 64;
#pragma unroll
    for (int i = 0; i < 16; ++i) {
      float4 v = src[i];
      u32x2 pk = { pk_bf16(v.x, v.y), pk_bf16(v.z, v.w) };
      *(u32x2*)(dst + i * 4) = pk;            // 8B aligned (XK*2=272, 128, 8)
    }
  }
  // ---- W1^T -> LDS: float4 reads, transposed b16 stores ----
  {
    const float* w1p = W1 + (size_t)e * (DDIM * HDIM);
    const int g = (tid & 63) * 4;             // h_out group
    const int doff = tid >> 6;                // 0..3
#pragma unroll 4
    for (int i = 0; i < DDIM / 4; ++i) {
      const int d = i * 4 + doff;
      float4 v = *(const float4*)(w1p + d * HDIM + g);
      st_bf16_hi(&WS[(g + 0) * WK1 + d], v.x);
      st_bf16_hi(&WS[(g + 1) * WK1 + d], v.y);
      st_bf16_hi(&WS[(g + 2) * WK1 + d], v.z);
      st_bf16_hi(&WS[(g + 3) * WK1 + d], v.w);
    }
  }
  // ---- prefetch W2 expert block toward L2 during layer 1 ----
  {
    const float* w2p = W2 + ((size_t)e << 16);
#pragma unroll
    for (int i = 0; i < 8; ++i)
      __builtin_prefetch(w2p + (size_t)(tid + i * 256) * 32, 0, 1);
  }
  __syncthreads();

  const int wv = tid >> 5, lane = tid & 31;
  const int ln = lane & 15, sel = lane >> 4;

  // ---- layer 1: [128x128] @ [128x256] (+b1, relu) -> H1 (bf16) ----
  {
    v16bf a[4];
    const int arow = wv * 16 + ln;
#pragma unroll
    for (int ks = 0; ks < 4; ++ks)
      a[ks] = load_frag_a(XS, arow, XK, ks * 32, sel * 8);
    const float* bp = b1 + e * HDIM;
    for (int ct = 0; ct < 16; ++ct) {
      const int n = ct * 16 + ln;
      v8f acc = {0.f, 0.f, 0.f, 0.f, 0.f, 0.f, 0.f, 0.f};   // inline-0 SRC2
#pragma unroll
      for (int ks = 0; ks < 4; ++ks) {
        v16bf bb = load_frag_b(WS, n, WK1, ks * 32, sel * 16);
        acc = wmma_bf16(a[ks], bb, acc);
      }
      const float bias = bp[n];
#pragma unroll
      for (int r = 0; r < 8; ++r)   // bias pairs -> v_pk_add_f32; relu -> v_med3
        st_bf16_hi(&H1[(wv * 16 + r + sel * 8) * HK + n], relu1(acc[r] + bias));
    }
  }
  __syncthreads();
  // ---- W2^T -> LDS (float4 reads); W3 -> row 0 of XS (reused as W3^T) ----
  {
    const float* w2p = W2 + ((size_t)e << 16);
    const int g = (tid & 63) * 4;             // h_out group
    const int hoff = tid >> 6;                // 0..3
#pragma unroll 4
    for (int i = 0; i < HDIM / 4; ++i) {
      const int h = i * 4 + hoff;
      float4 v = *(const float4*)(w2p + h * HDIM + g);
      st_bf16_hi(&WS[(g + 0) * WK2 + h], v.x);
      st_bf16_hi(&WS[(g + 1) * WK2 + h], v.y);
      st_bf16_hi(&WS[(g + 2) * WK2 + h], v.z);
      st_bf16_hi(&WS[(g + 3) * WK2 + h], v.w);
    }
    // W3^T row 0 (cols 1..15 of the tile are never read -> no zeroing needed)
    st_bf16_hi(&XS[tid], W3[e * HDIM + tid]);
  }
  __syncthreads();
  // ---- layer 2: [128x256] @ [256x256] (+b2, relu) -> H2 (bf16) ----
  {
    v16bf a[8];
    const int arow = wv * 16 + ln;
#pragma unroll
    for (int ks = 0; ks < 8; ++ks)
      a[ks] = load_frag_a(H1, arow, HK, ks * 32, sel * 8);
    const float* bp = b2 + e * HDIM;
    for (int ct = 0; ct < 16; ++ct) {
      const int n = ct * 16 + ln;
      v8f acc = {0.f, 0.f, 0.f, 0.f, 0.f, 0.f, 0.f, 0.f};
#pragma unroll
      for (int ks = 0; ks < 8; ++ks) {
        v16bf bb = load_frag_b(WS, n, WK2, ks * 32, sel * 16);
        acc = wmma_bf16(a[ks], bb, acc);
      }
      const float bias = bp[n];
#pragma unroll
      for (int r = 0; r < 8; ++r)
        st_bf16_hi(&H2[(wv * 16 + r + sel * 8) * HK + n], relu1(acc[r] + bias));
    }
  }
  __syncthreads();
  // ---- layer 3 via WMMA: B = W3^T tile (row 0 valid), read column N=0 ----
  {
    v16bf a[8];
    const int arow = wv * 16 + ln;
#pragma unroll
    for (int ks = 0; ks < 8; ++ks)
      a[ks] = load_frag_a(H2, arow, HK, ks * 32, sel * 8);
    v8f acc = {0.f, 0.f, 0.f, 0.f, 0.f, 0.f, 0.f, 0.f};
#pragma unroll
    for (int ks = 0; ks < 8; ++ks) {
      v16bf bb = load_frag_b(XS, ln, WK2, ks * 32, sel * 16);
      acc = wmma_bf16(a[ks], bb, acc);
    }
    if (ln == 0) {                            // lanes 0 & 16 hold column N=0
      const float b3e = b3[e];
#pragma unroll
      for (int r = 0; r < 8; ++r) {
        const int row = wv * 16 + r + sel * 8;
        if (lt * TILE_M + row < cnt) {
          const int tok = idxbuf[slotBase + row];
          out[tok] = acc[r] + b3e;
        }
      }
    }
  }
}

// ---------------- host-side launcher ----------------
extern "C" void kernel_launch(void* const* d_in, const int* in_sizes, int n_in,
                              void* d_out, int out_size, void* d_ws, size_t ws_size,
                              hipStream_t stream) {
  const float* x  = (const float*)d_in[0];
  const float* Wg = (const float*)d_in[1];
  const float* bg = (const float*)d_in[2];
  const float* W1 = (const float*)d_in[3];
  const float* b1 = (const float*)d_in[4];
  const float* W2 = (const float*)d_in[5];
  const float* b2 = (const float*)d_in[6];
  const float* W3 = (const float*)d_in[7];
  const float* b3 = (const float*)d_in[8];
  float* out = (float*)d_out;
  int* wsi = (int*)d_ws;

  moe_init_kernel<<<1, 64, 0, stream>>>(wsi);
  moe_gate_kernel<<<NTOK / 256, 256, 0, stream>>>(x, Wg, bg, wsi);
  moe_prefix_kernel<<<1, 32, 0, stream>>>(wsi);
  moe_scatter_kernel<<<NTOK / 256, 256, 0, stream>>>(wsi);

  const size_t smem = (size_t)SMEM_HALVES * sizeof(unsigned short); // ~298 KB
  moe_mlp_kernel<<<MAXTILES, 256, smem, stream>>>(x, W1, b1, W2, b2, W3, b3,
                                                  wsi, out);
}
